// dual_extraction_30219389895252
// MI455X (gfx1250) — matrix operand order repaired and checked by reading, not compiled, not run on hardware
//
#include <hip/hip_runtime.h>
#include <hip/hip_fp16.h>
#include <hip/hip_bf16.h>

// ---------------------------------------------------------------------------
// dual_extraction for MI455X (gfx1250, wave32, WMMA 16x16x32 f16->f32).
// Global NxN attention GEMMs run on WMMA; att matrix kept in f16 (fits L2).
// ---------------------------------------------------------------------------

#define B_ 2
#define N_ 4096
#define K_ 16

typedef __attribute__((ext_vector_type(16))) _Float16 v16h;
typedef __attribute__((ext_vector_type(8)))  _Float16 v8h;
typedef __attribute__((ext_vector_type(8)))  float    v8f;

static __device__ __forceinline__ v16h cat16(v8h lo, v8h hi) {
  return __builtin_shufflevector(lo, hi, 0,1,2,3,4,5,6,7,8,9,10,11,12,13,14,15);
}

// ---------------------------------------------------------------------------
// KNN: per point keep 17 nearest (squared L2, same formula as reference),
// drop nearest (self). idx layout [B][K][N].
// ---------------------------------------------------------------------------
__global__ void knn_k(const float* __restrict__ x, int* __restrict__ idx) {
  int t = blockIdx.x * blockDim.x + threadIdx.x;
  if (t >= B_ * N_) return;
  int b = t / N_, n = t % N_;
  const float* xb = x + (size_t)b * 3 * N_;
  float x0 = xb[n], x1 = xb[N_ + n], x2 = xb[2 * N_ + n];
  float sn = x0 * x0 + x1 * x1 + x2 * x2;
  float bd[17];
  int   bi[17];
#pragma unroll
  for (int i = 0; i < 17; ++i) { bd[i] = 1e30f; bi[i] = 0; }
  for (int m = 0; m < N_; ++m) {
    float y0 = xb[m], y1 = xb[N_ + m], y2 = xb[2 * N_ + m];
    float sm = y0 * y0 + y1 * y1 + y2 * y2;
    float d = sn + sm - 2.0f * (x0 * y0 + x1 * y1 + x2 * y2);
    if (d < bd[16]) {
      int p = 16;
      while (p > 0 && bd[p - 1] > d) { bd[p] = bd[p - 1]; bi[p] = bi[p - 1]; --p; }
      bd[p] = d; bi[p] = m;
    }
  }
  for (int kk = 0; kk < K_; ++kk) idx[((size_t)b * K_ + kk) * N_ + n] = bi[kk + 1];
}

// ---------------------------------------------------------------------------
// Generic pointwise conv (1x1): out[b][o][n] = W[o][:]·in[b][:][n] + bias[o]
// ---------------------------------------------------------------------------
__global__ void conv1d_k(const float* __restrict__ in, const float* __restrict__ W,
                         const float* __restrict__ bias, float* __restrict__ out,
                         int Ci, int Co) {
  int t = blockIdx.x * blockDim.x + threadIdx.x;
  int total = B_ * Co * N_;
  if (t >= total) return;
  int n = t % N_;
  int o = (t / N_) % Co;
  int b = t / (N_ * Co);
  const float* ib = in + (size_t)b * Ci * N_ + n;
  const float* w = W + (size_t)o * Ci;
  float acc = bias ? bias[o] : 0.0f;
  for (int i = 0; i < Ci; ++i) acc = fmaf(w[i], ib[(size_t)i * N_], acc);
  out[t] = acc;
}

// ---------------------------------------------------------------------------
// fea_fuse grouped convs (store conv output [B][Co][K][N] for BN stats)
// ---------------------------------------------------------------------------
__global__ void geo_conv_k(const float* __restrict__ xyz, const int* __restrict__ idx,
                           const float* __restrict__ W, float* __restrict__ out, int Co) {
  int t = blockIdx.x * blockDim.x + threadIdx.x;
  int total = B_ * Co * K_ * N_;
  if (t >= total) return;
  int n = t % N_;
  int kk = (t / N_) % K_;
  int o = (t / (N_ * K_)) % Co;
  int b = t / (N_ * K_ * Co);
  const float* xb = xyz + (size_t)b * 3 * N_;
  int j = idx[((size_t)b * K_ + kk) * N_ + n];
  const float* w = W + (size_t)o * 6;
  float acc = 0.0f;
#pragma unroll
  for (int c = 0; c < 3; ++c) {
    float cen = xb[(size_t)c * N_ + n];
    float nbr = xb[(size_t)c * N_ + j];
    acc = fmaf(w[c], cen, acc);
    acc = fmaf(w[3 + c], nbr - cen, acc);
  }
  out[t] = acc;
}

__global__ void ff_conv_k(const float* __restrict__ fea, const int* __restrict__ idx,
                          const float* __restrict__ W, float* __restrict__ out,
                          int Ci, int Co) {
  int t = blockIdx.x * blockDim.x + threadIdx.x;
  int total = B_ * Co * K_ * N_;
  if (t >= total) return;
  int n = t % N_;
  int kk = (t / N_) % K_;
  int o = (t / (N_ * K_)) % Co;
  int b = t / (N_ * K_ * Co);
  int j = idx[((size_t)b * K_ + kk) * N_ + n];
  const float* w = W + (size_t)o * (2 * Ci);
  float acc = 0.0f;
  for (int c = 0; c < Ci; ++c) {
    float fc = fea[((size_t)b * Ci + c) * N_ + n];
    float fn = fea[((size_t)b * Ci + c) * N_ + j];
    acc = fmaf(w[c], fc, acc);
    acc = fmaf(w[Ci + c], fn - fc, acc);
  }
  out[t] = acc;
}

// ---------------------------------------------------------------------------
// Reductions for training-mode BN: stats[2c]=sum, stats[2c+1]=sumsq
// in layout [B][C][M] (M = K*N or N)
// ---------------------------------------------------------------------------
__global__ void zero_k(float* __restrict__ p, int n) {
  int i = blockIdx.x * blockDim.x + threadIdx.x;
  if (i < n) p[i] = 0.0f;
}

__global__ void stats_k(const float* __restrict__ in, float* __restrict__ stats,
                        int C, int M) {
  int c = blockIdx.y;
  int tid = blockIdx.x * blockDim.x + threadIdx.x;
  int stride = gridDim.x * blockDim.x;
  int total = B_ * M;
  float s = 0.0f, s2 = 0.0f;
  for (int l = tid; l < total; l += stride) {
    int b = l / M, m = l % M;
    float v = in[((size_t)b * C + c) * M + m];
    s += v; s2 += v * v;
  }
  __shared__ float sh1[256], sh2[256];
  sh1[threadIdx.x] = s; sh2[threadIdx.x] = s2;
  __syncthreads();
  for (int o = 128; o > 0; o >>= 1) {
    if ((int)threadIdx.x < o) { sh1[threadIdx.x] += sh1[threadIdx.x + o]; sh2[threadIdx.x] += sh2[threadIdx.x + o]; }
    __syncthreads();
  }
  if (threadIdx.x == 0) {
    atomicAdd(&stats[2 * c], sh1[0]);
    atomicAdd(&stats[2 * c + 1], sh2[0]);
  }
}

// BN + relu + max over K: out[b][Ooff+o][n] (Ostride total channels)
__global__ void bn_relu_max_k(const float* __restrict__ conv, const float* __restrict__ stats,
                              const float* __restrict__ g, const float* __restrict__ beta,
                              float* __restrict__ out, int Co, int Ooff, int Ostride,
                              float inv_cnt) {
  int t = blockIdx.x * blockDim.x + threadIdx.x;
  int total = B_ * Co * N_;
  if (t >= total) return;
  int n = t % N_;
  int o = (t / N_) % Co;
  int b = t / (N_ * Co);
  float mean = stats[2 * o] * inv_cnt;
  float var = stats[2 * o + 1] * inv_cnt - mean * mean;
  float sc = g[o] * rsqrtf(var + 1e-5f);
  float sh = beta[o] - sc * mean;
  float m = 0.0f;  // relu(.) >= 0, so init 0 gives relu+max combined
#pragma unroll
  for (int kk = 0; kk < K_; ++kk) {
    float v = fmaf(sc, conv[(((size_t)b * Co + o) * K_ + kk) * N_ + n], sh);
    m = fmaxf(m, v);
  }
  out[((size_t)b * Ostride + Ooff + o) * N_ + n] = m;
}

// BN (axes (0,2)) + relu, layout [B][C][N], in-place capable
__global__ void bn_relu_k(const float* __restrict__ in, const float* __restrict__ stats,
                          const float* __restrict__ g, const float* __restrict__ beta,
                          float* __restrict__ out, int C, float inv_cnt) {
  int t = blockIdx.x * blockDim.x + threadIdx.x;
  int total = B_ * C * N_;
  if (t >= total) return;
  int o = (t / N_) % C;
  float mean = stats[2 * o] * inv_cnt;
  float var = stats[2 * o + 1] * inv_cnt - mean * mean;
  float sc = g[o] * rsqrtf(var + 1e-5f);
  float sh = beta[o] - sc * mean;
  out[t] = fmaxf(fmaf(sc, in[t], sh), 0.0f);
}

// ---------------------------------------------------------------------------
// f16 staging for WMMA: qt/kt transposed [B][N][64], vh = [B][80][N]
// (rows 0..63 = v, row 64 = ones so colsum falls out of the GEMM, 65..79 = 0)
// ---------------------------------------------------------------------------
__global__ void cvt_tr16_k(const float* __restrict__ src, _Float16* __restrict__ dst) {
  int t = blockIdx.x * blockDim.x + threadIdx.x;
  int total = B_ * N_ * 64;
  if (t >= total) return;
  int d = t % 64;
  int n = (t / 64) % N_;
  int b = t / (64 * N_);
  dst[t] = (_Float16)src[((size_t)b * 64 + d) * N_ + n];
}

__global__ void build_vh_k(const float* __restrict__ v, _Float16* __restrict__ vh) {
  int t = blockIdx.x * blockDim.x + threadIdx.x;
  int total = B_ * 80 * N_;
  if (t >= total) return;
  int n = t % N_;
  int dd = (t / N_) % 80;
  int b = t / (80 * N_);
  float val = (dd < 64) ? v[((size_t)b * 64 + dd) * N_ + n] : ((dd == 64) ? 1.0f : 0.0f);
  vh[t] = (_Float16)val;
}

// ---------------------------------------------------------------------------
// Attention pass 1: online row-softmax stats of energy = q^T k via WMMA.
// One wave handles (b, ntile, mchunk of 32 mtiles). Partial (max,sum) -> ws.
// ---------------------------------------------------------------------------
__global__ void attn_pass1_k(const _Float16* __restrict__ qt, const _Float16* __restrict__ kt,
                             float* __restrict__ pmax, float* __restrict__ psum) {
  int wid = (blockIdx.x * blockDim.x + threadIdx.x) >> 5;
  int lane = threadIdx.x & 31;
  int chunk = wid & 7;
  int ntile = (wid >> 3) & 255;
  int b = wid >> 11;
  int n0 = ntile * 16;
  int col = lane & 15, hi = lane >> 4;
  const _Float16* qtb = qt + (size_t)b * N_ * 64;
  const _Float16* ktb = kt + (size_t)b * N_ * 64;
  // A fragments (q rows = points n0..n0+15), both K-steps, reused over mtiles
  const _Float16* ab = qtb + (size_t)(n0 + col) * 64 + hi * 8;
  v16h A0 = cat16(*(const v8h*)(ab), *(const v8h*)(ab + 16));
  v16h A1 = cat16(*(const v8h*)(ab + 32), *(const v8h*)(ab + 48));
  float runmax[8], runsum[8];
#pragma unroll
  for (int r = 0; r < 8; ++r) { runmax[r] = -1e30f; runsum[r] = 0.0f; }
  for (int mt = chunk * 32; mt < chunk * 32 + 32; ++mt) {
    int m0 = mt * 16;
    const _Float16* kb = ktb + (size_t)(m0 + col) * 64 + hi * 8;
    v16h B0 = cat16(*(const v8h*)(kb), *(const v8h*)(kb + 16));
    v16h B1 = cat16(*(const v8h*)(kb + 32), *(const v8h*)(kb + 48));
    v8f c = {0.f, 0.f, 0.f, 0.f, 0.f, 0.f, 0.f, 0.f};
    c = __builtin_amdgcn_wmma_f32_16x16x32_f16(false, A0, false, B0, (short)0, c, false, false);
    c = __builtin_amdgcn_wmma_f32_16x16x32_f16(false, A1, false, B1, (short)0, c, false, false);
#pragma unroll
    for (int r = 0; r < 8; ++r) {
      float e = c[r];
      float tm = e;
      tm = fmaxf(tm, __shfl_xor(tm, 1));
      tm = fmaxf(tm, __shfl_xor(tm, 2));
      tm = fmaxf(tm, __shfl_xor(tm, 4));
      tm = fmaxf(tm, __shfl_xor(tm, 8));
      float nm = fmaxf(runmax[r], tm);
      float ex = __expf(e - nm);
      ex += __shfl_xor(ex, 1);
      ex += __shfl_xor(ex, 2);
      ex += __shfl_xor(ex, 4);
      ex += __shfl_xor(ex, 8);
      runsum[r] = runsum[r] * __expf(runmax[r] - nm) + ex;
      runmax[r] = nm;
    }
  }
  if (col == 0) {
#pragma unroll
    for (int r = 0; r < 8; ++r) {
      int row = n0 + hi * 8 + r;
      pmax[((size_t)b * N_ + row) * 8 + chunk] = runmax[r];
      psum[((size_t)b * N_ + row) * 8 + chunk] = runsum[r];
    }
  }
}

__global__ void attn_combine_k(const float* __restrict__ pmax, const float* __restrict__ psum,
                               float* __restrict__ rowmax, float* __restrict__ rowsum) {
  int t = blockIdx.x * blockDim.x + threadIdx.x;
  if (t >= B_ * N_) return;
  float M = -1e30f;
#pragma unroll
  for (int i = 0; i < 8; ++i) M = fmaxf(M, pmax[(size_t)t * 8 + i]);
  float S = 0.0f;
#pragma unroll
  for (int i = 0; i < 8; ++i) S += psum[(size_t)t * 8 + i] * __expf(pmax[(size_t)t * 8 + i] - M);
  rowmax[t] = M;
  rowsum[t] = S;
}

// ---------------------------------------------------------------------------
// Attention pass 2: materialize att (row softmax) as f16, transposed [m][n]
// so subsequent WMMA B-fragments are contiguous. Each lane: one v8h store.
// ---------------------------------------------------------------------------
__global__ void attn_att_k(const _Float16* __restrict__ qt, const _Float16* __restrict__ kt,
                           const float* __restrict__ rowmax, const float* __restrict__ rowsum,
                           _Float16* __restrict__ att_t, int b) {
  int wid = (blockIdx.x * blockDim.x + threadIdx.x) >> 5;
  int lane = threadIdx.x & 31;
  int mt = wid & 255;
  int nt = wid >> 8;
  int n0 = nt * 16, m0 = mt * 16;
  int col = lane & 15, hi = lane >> 4;
  const _Float16* qtb = qt + (size_t)b * N_ * 64;
  const _Float16* ktb = kt + (size_t)b * N_ * 64;
  const _Float16* ab = qtb + (size_t)(n0 + col) * 64 + hi * 8;
  v16h A0 = cat16(*(const v8h*)(ab), *(const v8h*)(ab + 16));
  v16h A1 = cat16(*(const v8h*)(ab + 32), *(const v8h*)(ab + 48));
  const _Float16* kb = ktb + (size_t)(m0 + col) * 64 + hi * 8;
  v16h B0 = cat16(*(const v8h*)(kb), *(const v8h*)(kb + 16));
  v16h B1 = cat16(*(const v8h*)(kb + 32), *(const v8h*)(kb + 48));
  v8f c = {0.f, 0.f, 0.f, 0.f, 0.f, 0.f, 0.f, 0.f};
  c = __builtin_amdgcn_wmma_f32_16x16x32_f16(false, A0, false, B0, (short)0, c, false, false);
  c = __builtin_amdgcn_wmma_f32_16x16x32_f16(false, A1, false, B1, (short)0, c, false, false);
  const float* rm = rowmax + (size_t)b * N_;
  const float* rs = rowsum + (size_t)b * N_;
  v8h o;
#pragma unroll
  for (int r = 0; r < 8; ++r) {
    int row = n0 + hi * 8 + r;
    float a = __expf(c[r] - rm[row]) / rs[row];
    o[r] = (_Float16)a;
  }
  *(v8h*)(att_t + (size_t)(m0 + col) * N_ + n0 + hi * 8) = o;
}

// ---------------------------------------------------------------------------
// numer[dd][m] = sum_n vh[dd][n] * att[n][m], dd in [0,80): row 64 = colsum.
// Each wave: (mtile, nchunk of 256), 5 accumulators, atomicAdd f32 partials.
// ---------------------------------------------------------------------------
__global__ void attn_numer_k(const _Float16* __restrict__ vh, const _Float16* __restrict__ att_t,
                             float* __restrict__ numer, int b) {
  int wid = (blockIdx.x * blockDim.x + threadIdx.x) >> 5;
  int lane = threadIdx.x & 31;
  int mt = wid & 255;
  int nc = wid >> 8;
  int m0 = mt * 16;
  int col = lane & 15, hi = lane >> 4;
  const _Float16* vb = vh + (size_t)b * 80 * N_;
  v8f acc[5];
#pragma unroll
  for (int dt = 0; dt < 5; ++dt) acc[dt] = (v8f){0.f, 0.f, 0.f, 0.f, 0.f, 0.f, 0.f, 0.f};
  for (int ks = 0; ks < 8; ++ks) {
    int nb = nc * 256 + ks * 32;
    const _Float16* bp = att_t + (size_t)(m0 + col) * N_ + nb + hi * 8;
    v16h Bf = cat16(*(const v8h*)(bp), *(const v8h*)(bp + 16));
#pragma unroll
    for (int dt = 0; dt < 5; ++dt) {
      const _Float16* ap = vb + (size_t)(dt * 16 + col) * N_ + nb + hi * 8;
      v16h Af = cat16(*(const v8h*)(ap), *(const v8h*)(ap + 16));
      acc[dt] = __builtin_amdgcn_wmma_f32_16x16x32_f16(false, Af, false, Bf, (short)0, acc[dt], false, false);
    }
  }
  float* nout = numer + (size_t)b * 80 * N_;
#pragma unroll
  for (int dt = 0; dt < 5; ++dt)
#pragma unroll
    for (int r = 0; r < 8; ++r)
      atomicAdd(&nout[(size_t)(dt * 16 + hi * 8 + r) * N_ + m0 + col], acc[dt][r]);
}

// x - x_r, with x_r = numer / (1e-9 + colsum)
__global__ void sub_xr_k(const float* __restrict__ x, const float* __restrict__ numer,
                         float* __restrict__ out) {
  int t = blockIdx.x * blockDim.x + threadIdx.x;
  int total = B_ * 64 * N_;
  if (t >= total) return;
  int n = t % N_;
  int d = (t / N_) % 64;
  int b = t / (64 * N_);
  float cs = numer[((size_t)b * 80 + 64) * N_ + n];
  float xr = numer[((size_t)b * 80 + d) * N_ + n] / (1e-9f + cs);
  out[t] = x[t] - xr;
}

__global__ void add_k(const float* __restrict__ a, const float* __restrict__ b,
                      float* __restrict__ o, int total) {
  int t = blockIdx.x * blockDim.x + threadIdx.x;
  if (t < total) o[t] = a[t] + b[t];
}

// ---------------------------------------------------------------------------
// Local (K=16) point-transformer branch, one block per point, fused:
// pos = d2(relu(d1(rel_pos))); a = g2(relu(g1(q - k_nbr + pos)));
// softmax over K; res = sum_k a*(v_nbr + pos)
// ---------------------------------------------------------------------------
__global__ void local_attn_k(const float* __restrict__ xyz, const int* __restrict__ idx,
                             const float* __restrict__ q, const float* __restrict__ kf,
                             const float* __restrict__ vf,
                             const float* __restrict__ d1w, const float* __restrict__ d1b,
                             const float* __restrict__ d2w, const float* __restrict__ d2b,
                             const float* __restrict__ g1w, const float* __restrict__ g1b,
                             const float* __restrict__ g2w, const float* __restrict__ g2b,
                             float* __restrict__ res) {
  __shared__ float rp[10];
  __shared__ float h1[64];
  __shared__ float a0[64];
  __shared__ float h256[256];
  __shared__ float gpart[256];
  __shared__ float aatt[64 * K_];
  __shared__ float wsum[64 * K_];
  int b = blockIdx.x / N_;
  int n = blockIdx.x % N_;
  int t = threadIdx.x;
  const float* xb = xyz + (size_t)b * 3 * N_;
  for (int kk = 0; kk < K_; ++kk) {
    int j = idx[((size_t)b * K_ + kk) * N_ + n];
    if (t < 10) {
      float val;
      if (t < 3) val = xb[(size_t)t * N_ + n];
      else if (t < 6) val = xb[(size_t)(t - 3) * N_ + j];
      else if (t < 9) val = xb[(size_t)(t - 6) * N_ + j] - xb[(size_t)(t - 6) * N_ + n];
      else {
        float r0 = xb[n] - xb[j];
        float r1 = xb[N_ + n] - xb[N_ + j];
        float r2 = xb[2 * N_ + n] - xb[2 * N_ + j];
        val = sqrtf(r0 * r0 + r1 * r1 + r2 * r2);
      }
      rp[t] = val;
    }
    __syncthreads();
    if (t < 64) {
      float a = d1b[t];
#pragma unroll
      for (int c = 0; c < 10; ++c) a = fmaf(d1w[t * 10 + c], rp[c], a);
      h1[t] = fmaxf(a, 0.0f);
    }
    __syncthreads();
    if (t < 64) {
      float a = d2b[t];
      for (int o2 = 0; o2 < 64; ++o2) a = fmaf(d2w[t * 64 + o2], h1[o2], a);
      float qv = q[((size_t)b * 64 + t) * N_ + n];
      float kv = kf[((size_t)b * 64 + t) * N_ + j];
      float vv = vf[((size_t)b * 64 + t) * N_ + j];
      a0[t] = qv - kv + a;
      wsum[t * K_ + kk] = vv + a;
    }
    __syncthreads();
    {
      float a = g1b[t];
      for (int d = 0; d < 64; ++d) a = fmaf(g1w[t * 64 + d], a0[d], a);
      h256[t] = fmaxf(a, 0.0f);
    }
    __syncthreads();
    {
      int d = t & 63, part = t >> 6;
      int base = part * 64;
      float a = 0.0f;
      for (int i = 0; i < 64; ++i) a = fmaf(g2w[d * 256 + base + i], h256[base + i], a);
      gpart[t] = a;
    }
    __syncthreads();
    if (t < 64)
      aatt[t * K_ + kk] = g2b[t] + gpart[t] + gpart[64 + t] + gpart[128 + t] + gpart[192 + t];
    __syncthreads();
  }
  if (t < 64) {
    float mx = -1e30f;
#pragma unroll
    for (int kk = 0; kk < K_; ++kk) mx = fmaxf(mx, aatt[t * K_ + kk]);
    float e[K_];
    float s = 0.0f;
#pragma unroll
    for (int kk = 0; kk < K_; ++kk) { e[kk] = __expf(aatt[t * K_ + kk] - mx); s += e[kk]; }
    float inv = 1.0f / s;
    float r = 0.0f;
#pragma unroll
    for (int kk = 0; kk < K_; ++kk) r = fmaf(e[kk] * inv, wsum[t * K_ + kk], r);
    res[((size_t)b * 64 + t) * N_ + n] = r;
  }
}

// ---------------------------------------------------------------------------
// Host side
// ---------------------------------------------------------------------------
struct DTP {
  const float *c1_w, *bn1_g, *bn1_b, *c2f_w, *bn2_g, *bn2_b;
  const float *conv1_w, *conv1_b, *wq_w, *wq_b, *wk_w, *wk_b, *wv_w, *wv_b;
  const float *d1_w, *d1_b, *d2_w, *d2_b, *g1_w, *g1_b, *g2_w, *g2_b;
  const float *t_w, *t_b, *an_g, *an_b, *c2_w, *c2_b, *c3_w, *c3_b;
  const float *l_w, *l_b, *lbn_g, *lbn_b;
};

struct WSP {
  int* idx;
  float *fea0, *convKN, *stats, *ffout;
  float *x, *q, *kbuf, *vbuf;
  _Float16 *qt, *kt, *vh;
  float *pmax, *psum, *rowmax, *rowsum;
  _Float16* att_t;
  float *numer, *t0, *t1, *resp, *resg, *resl, *lt, *dt1out;
};

static DTP load_dtp(void* const* d_in, int base) {
  DTP p;
  int i = base;
  p.c1_w = (const float*)d_in[i++];  p.bn1_g = (const float*)d_in[i++]; p.bn1_b = (const float*)d_in[i++];
  p.c2f_w = (const float*)d_in[i++]; p.bn2_g = (const float*)d_in[i++]; p.bn2_b = (const float*)d_in[i++];
  p.conv1_w = (const float*)d_in[i++]; p.conv1_b = (const float*)d_in[i++];
  p.wq_w = (const float*)d_in[i++]; p.wq_b = (const float*)d_in[i++];
  p.wk_w = (const float*)d_in[i++]; p.wk_b = (const float*)d_in[i++];
  p.wv_w = (const float*)d_in[i++]; p.wv_b = (const float*)d_in[i++];
  p.d1_w = (const float*)d_in[i++]; p.d1_b = (const float*)d_in[i++];
  p.d2_w = (const float*)d_in[i++]; p.d2_b = (const float*)d_in[i++];
  p.g1_w = (const float*)d_in[i++]; p.g1_b = (const float*)d_in[i++];
  p.g2_w = (const float*)d_in[i++]; p.g2_b = (const float*)d_in[i++];
  p.t_w = (const float*)d_in[i++]; p.t_b = (const float*)d_in[i++];
  p.an_g = (const float*)d_in[i++]; p.an_b = (const float*)d_in[i++];
  p.c2_w = (const float*)d_in[i++]; p.c2_b = (const float*)d_in[i++];
  p.c3_w = (const float*)d_in[i++]; p.c3_b = (const float*)d_in[i++];
  p.l_w = (const float*)d_in[i++]; p.l_b = (const float*)d_in[i++];
  p.lbn_g = (const float*)d_in[i++]; p.lbn_b = (const float*)d_in[i++];
  return p;
}

static inline dim3 blocks_for(int total) { return dim3((unsigned)((total + 255) / 256)); }

static void run_dt(const DTP& p, const float* xyz, const int* idx, const float* fea_in,
                   int in_c, int out_c, float* out, const WSP& w, hipStream_t s) {
  const dim3 blk(256);
  const int oc2 = 2 * out_c;
  const float inv_bkn = 1.0f / (float)(B_ * K_ * N_);
  const float inv_bn = 1.0f / (float)(B_ * N_);

  // fea_fuse geometric part
  geo_conv_k<<<blocks_for(B_ * out_c * K_ * N_), blk, 0, s>>>(xyz, idx, p.c1_w, w.convKN, out_c);
  zero_k<<<blocks_for(256), blk, 0, s>>>(w.stats, 256);
  stats_k<<<dim3(32, out_c), blk, 0, s>>>(w.convKN, w.stats, out_c, K_ * N_);
  bn_relu_max_k<<<blocks_for(B_ * out_c * N_), blk, 0, s>>>(w.convKN, w.stats, p.bn1_g, p.bn1_b,
                                                            w.ffout, out_c, 0, oc2, inv_bkn);
  // fea_fuse feature part
  ff_conv_k<<<blocks_for(B_ * out_c * K_ * N_), blk, 0, s>>>(fea_in, idx, p.c2f_w, w.convKN, in_c, out_c);
  zero_k<<<blocks_for(256), blk, 0, s>>>(w.stats, 256);
  stats_k<<<dim3(32, out_c), blk, 0, s>>>(w.convKN, w.stats, out_c, K_ * N_);
  bn_relu_max_k<<<blocks_for(B_ * out_c * N_), blk, 0, s>>>(w.convKN, w.stats, p.bn2_g, p.bn2_b,
                                                            w.ffout, out_c, out_c, oc2, inv_bkn);
  // x, q, k, v
  conv1d_k<<<blocks_for(B_ * 64 * N_), blk, 0, s>>>(w.ffout, p.conv1_w, p.conv1_b, w.x, oc2, 64);
  conv1d_k<<<blocks_for(B_ * 64 * N_), blk, 0, s>>>(w.x, p.wq_w, p.wq_b, w.q, 64, 64);
  conv1d_k<<<blocks_for(B_ * 64 * N_), blk, 0, s>>>(w.x, p.wk_w, p.wk_b, w.kbuf, 64, 64);
  conv1d_k<<<blocks_for(B_ * 64 * N_), blk, 0, s>>>(w.x, p.wv_w, p.wv_b, w.vbuf, 64, 64);
  // f16 staging
  cvt_tr16_k<<<blocks_for(B_ * N_ * 64), blk, 0, s>>>(w.q, w.qt);
  cvt_tr16_k<<<blocks_for(B_ * N_ * 64), blk, 0, s>>>(w.kbuf, w.kt);
  build_vh_k<<<blocks_for(B_ * 80 * N_), blk, 0, s>>>(w.vbuf, w.vh);
  // global attention (WMMA)
  attn_pass1_k<<<dim3(512), blk, 0, s>>>(w.qt, w.kt, w.pmax, w.psum);
  attn_combine_k<<<blocks_for(B_ * N_), blk, 0, s>>>(w.pmax, w.psum, w.rowmax, w.rowsum);
  zero_k<<<blocks_for(B_ * 80 * N_), blk, 0, s>>>(w.numer, B_ * 80 * N_);
  for (int b = 0; b < B_; ++b) {
    attn_att_k<<<dim3(8192), blk, 0, s>>>(w.qt, w.kt, w.rowmax, w.rowsum, w.att_t, b);
    attn_numer_k<<<dim3(512), blk, 0, s>>>(w.vh, w.att_t, w.numer, b);
  }
  // global branch epilogue
  sub_xr_k<<<blocks_for(B_ * 64 * N_), blk, 0, s>>>(w.x, w.numer, w.t0);
  conv1d_k<<<blocks_for(B_ * 64 * N_), blk, 0, s>>>(w.t0, p.t_w, p.t_b, w.t1, 64, 64);
  zero_k<<<blocks_for(256), blk, 0, s>>>(w.stats, 256);
  stats_k<<<dim3(32, 64), blk, 0, s>>>(w.t1, w.stats, 64, N_);
  bn_relu_k<<<blocks_for(B_ * 64 * N_), blk, 0, s>>>(w.t1, w.stats, p.an_g, p.an_b, w.t1, 64, inv_bn);
  add_k<<<blocks_for(B_ * 64 * N_), blk, 0, s>>>(w.x, w.t1, w.t0, B_ * 64 * N_);
  conv1d_k<<<blocks_for(B_ * oc2 * N_), blk, 0, s>>>(w.t0, p.c3_w, p.c3_b, w.resg, 64, oc2);
  // local branch
  local_attn_k<<<dim3(B_ * N_), blk, 0, s>>>(xyz, idx, w.q, w.kbuf, w.vbuf,
                                             p.d1_w, p.d1_b, p.d2_w, p.d2_b,
                                             p.g1_w, p.g1_b, p.g2_w, p.g2_b, w.resp);
  conv1d_k<<<blocks_for(B_ * oc2 * N_), blk, 0, s>>>(w.resp, p.c2_w, p.c2_b, w.resl, 64, oc2);
  // fuse
  add_k<<<blocks_for(B_ * oc2 * N_), blk, 0, s>>>(w.resg, w.resl, w.resg, B_ * oc2 * N_);
  conv1d_k<<<blocks_for(B_ * oc2 * N_), blk, 0, s>>>(w.resg, p.l_w, p.l_b, w.lt, oc2, oc2);
  zero_k<<<blocks_for(256), blk, 0, s>>>(w.stats, 256);
  stats_k<<<dim3(32, oc2), blk, 0, s>>>(w.lt, w.stats, oc2, N_);
  bn_relu_k<<<blocks_for(B_ * oc2 * N_), blk, 0, s>>>(w.lt, w.stats, p.lbn_g, p.lbn_b, w.lt, oc2, inv_bn);
  add_k<<<blocks_for(B_ * oc2 * N_), blk, 0, s>>>(w.ffout, w.lt, out, B_ * oc2 * N_);
}

extern "C" void kernel_launch(void* const* d_in, const int* in_sizes, int n_in,
                              void* d_out, int out_size, void* d_ws, size_t ws_size,
                              hipStream_t stream) {
  (void)in_sizes; (void)n_in; (void)out_size; (void)ws_size;
  const float* x = (const float*)d_in[0];
  const float* conv_w = (const float*)d_in[1];
  const float* conv_b = (const float*)d_in[2];
  DTP dt1 = load_dtp(d_in, 3);
  DTP dt2 = load_dtp(d_in, 37);

  // arena (~115 MB total)
  char* base = (char*)d_ws;
  size_t off = 0;
  auto alloc = [&](size_t bytes) -> void* {
    void* p = base + off;
    off += (bytes + 255) & ~(size_t)255;
    return p;
  };
  WSP w;
  w.idx = (int*)alloc((size_t)B_ * K_ * N_ * 4);
  w.fea0 = (float*)alloc((size_t)B_ * 16 * N_ * 4);
  w.convKN = (float*)alloc((size_t)B_ * 64 * K_ * N_ * 4);
  w.stats = (float*)alloc(256 * 4);
  w.ffout = (float*)alloc((size_t)B_ * 128 * N_ * 4);
  w.x = (float*)alloc((size_t)B_ * 64 * N_ * 4);
  w.q = (float*)alloc((size_t)B_ * 64 * N_ * 4);
  w.kbuf = (float*)alloc((size_t)B_ * 64 * N_ * 4);
  w.vbuf = (float*)alloc((size_t)B_ * 64 * N_ * 4);
  w.qt = (_Float16*)alloc((size_t)B_ * N_ * 64 * 2);
  w.kt = (_Float16*)alloc((size_t)B_ * N_ * 64 * 2);
  w.vh = (_Float16*)alloc((size_t)B_ * 80 * N_ * 2);
  w.pmax = (float*)alloc((size_t)B_ * N_ * 8 * 4);
  w.psum = (float*)alloc((size_t)B_ * N_ * 8 * 4);
  w.rowmax = (float*)alloc((size_t)B_ * N_ * 4);
  w.rowsum = (float*)alloc((size_t)B_ * N_ * 4);
  w.att_t = (_Float16*)alloc((size_t)N_ * N_ * 2);
  w.numer = (float*)alloc((size_t)B_ * 80 * N_ * 4);
  w.t0 = (float*)alloc((size_t)B_ * 64 * N_ * 4);
  w.t1 = (float*)alloc((size_t)B_ * 64 * N_ * 4);
  w.resp = (float*)alloc((size_t)B_ * 64 * N_ * 4);
  w.resg = (float*)alloc((size_t)B_ * 128 * N_ * 4);
  w.resl = (float*)alloc((size_t)B_ * 128 * N_ * 4);
  w.lt = (float*)alloc((size_t)B_ * 128 * N_ * 4);
  w.dt1out = (float*)alloc((size_t)B_ * 64 * N_ * 4);

  const dim3 blk(256);
  knn_k<<<blocks_for(B_ * N_), blk, 0, stream>>>(x, w.idx);
  conv1d_k<<<blocks_for(B_ * 16 * N_), blk, 0, stream>>>(x, conv_w, conv_b, w.fea0, 3, 16);
  run_dt(dt1, x, w.idx, w.fea0, 16, 32, w.dt1out, w, stream);
  run_dt(dt2, x, w.idx, w.dt1out, 64, 64, (float*)d_out, w, stream);
}
